// GraphNeuralNetwork_56410100466219
// MI455X (gfx1250) — matrix operand order repaired and checked by reading, not compiled, not run on hardware
//
#include <hip/hip_runtime.h>
#include <hip/hip_bf16.h>
#include <math.h>

// ---------- types for CDNA5 WMMA ----------
typedef __attribute__((ext_vector_type(16))) __bf16 v16bf;
typedef __attribute__((ext_vector_type(8)))  float  v8f;

#define BM 128   // block M tile (8 waves x 16 rows)
#define BN 32    // block N tile (2 x 16-col WMMA tiles per wave)

// ---------- device helpers ----------
__device__ inline unsigned short f2bf(float f) {
  unsigned int u = __float_as_uint(f);
  unsigned int r = u + 0x7FFFu + ((u >> 16) & 1u);   // round-to-nearest-even
  return (unsigned short)(r >> 16);
}
// monotone float->uint encoding so unsigned atomicMax == float max
__device__ inline unsigned int fenc(float f) {
  unsigned int u = __float_as_uint(f);
  return (u & 0x80000000u) ? ~u : (u | 0x80000000u);
}
__device__ inline float fdec(unsigned int e) {
  unsigned int u = (e & 0x80000000u) ? (e ^ 0x80000000u) : ~e;
  return __uint_as_float(u);
}

// ---------- pack fp32 weights K x N -> bf16 WMMA-B fragment order ----------
// Per 32(K) x 16(N) tile, per column c: 16 contiguous dwords =
//   dwords 0..7  : K pairs (2j, 2j+1)        (lane<16 half,  K 0..15)
//   dwords 8..15 : K pairs (16+2j, 16+2j+1)  (lane>=16 half, K 16..31)
// Flat dword index: ((tk*ntiles + tn)*16 + c)*16 + hb*8 + j
__global__ void k_pack_w(const float* __restrict__ W, unsigned short* __restrict__ out,
                         int K, int N) {
  int t = blockIdx.x * blockDim.x + threadIdx.x;   // output dword index
  if (t >= (K * N) >> 1) return;
  int j    = t & 7;
  int hb   = (t >> 3) & 1;
  int c    = (t >> 4) & 15;
  int rest = t >> 8;
  int ntiles = N >> 4;
  int tn = rest % ntiles;
  int tk = rest / ntiles;
  int k = (tk << 5) + (hb << 4) + (j << 1);
  int n = (tn << 4) + c;
  unsigned int lo = f2bf(W[(size_t)k * N + n]);
  unsigned int hi = f2bf(W[(size_t)(k + 1) * N + n]);
  ((unsigned int*)out)[t] = lo | (hi << 16);
}

// ---------- WMMA bf16 GEMM: C[Mpad x N] = A[Mpad x K] * B[K x N] ----------
// A row-major bf16 staged through LDS; B pre-packed in fragment order (global).
// 256 threads = 8 waves; wave w owns rows m0+16w..+15 and computes 2 n-tiles.
__global__ __launch_bounds__(256) void k_gemm_bf16(
    const unsigned short* __restrict__ A,    // Mpad x K bf16
    const unsigned short* __restrict__ Bp,   // packed weights (k_pack_w)
    float* __restrict__ C, int K, int N) {
  __shared__ unsigned short As[BM][40];      // 80B row stride: 16B-aligned, bank-spread
  const int tid    = threadIdx.x;
  const int lane   = tid & 31;
  const int wv     = tid >> 5;
  const int m0     = blockIdx.x * BM;
  const int ntile0 = blockIdx.y * 2;
  const int ntiles = N >> 4;

  const int lrow = tid >> 1;                 // coop A load: 128 rows x 32 cols
  const int lseg = (tid & 1) * 16;           // 16 bf16 per thread (2 x b128)

  const int c15 = lane & 15;
  const int hb  = lane >> 4;                 // 0: K 0..15 half, 1: K 16..31 half
  const int am  = wv * 16 + c15;             // A fragment row in LDS tile

  v8f acc0 = {}, acc1 = {};
  for (int k0 = 0; k0 < K; k0 += 32) {
    const unsigned short* arow = A + (size_t)(m0 + lrow) * K + k0 + lseg;
    uint4 av0 = *(const uint4*)(arow);
    uint4 av1 = *(const uint4*)(arow + 8);
    // B fragments: 8 contiguous dwords per lane, straight from packed global
    const int tk = k0 >> 5;
    const uint4* b0p = (const uint4*)(Bp +
        ((((size_t)tk * ntiles + ntile0) * 16 + c15) * 16 + hb * 8) * 2);
    const uint4* b1p = (const uint4*)(Bp +
        ((((size_t)tk * ntiles + ntile0 + 1) * 16 + c15) * 16 + hb * 8) * 2);
    uint4 b0a = b0p[0], b0b = b0p[1];
    uint4 b1a = b1p[0], b1b = b1p[1];

    __syncthreads();
    *(uint4*)&As[lrow][lseg]     = av0;
    *(uint4*)&As[lrow][lseg + 8] = av1;
    __syncthreads();

    union U { uint4 q[2]; v16bf v; } fa, fb0, fb1;
    fa.q[0] = *(const uint4*)&As[am][hb * 16];       // K 16*hb..16*hb+7
    fa.q[1] = *(const uint4*)&As[am][hb * 16 + 8];   // K 16*hb+8..16*hb+15
    fb0.q[0] = b0a; fb0.q[1] = b0b;
    fb1.q[0] = b1a; fb1.q[1] = b1b;
    acc0 = __builtin_amdgcn_wmma_f32_16x16x32_bf16(
        false, fa.v, false, fb0.v, (short)0, acc0, false, false);
    acc1 = __builtin_amdgcn_wmma_f32_16x16x32_bf16(
        false, fa.v, false, fb1.v, (short)0, acc1, false, false);
  }
  // D layout: VGPR r -> row r (lanes 0-15) / row r+8 (lanes 16-31), col = lane&15
  const int col0 = (ntile0 << 4) + c15;
  const int rb   = m0 + wv * 16 + (hb ? 8 : 0);
#pragma unroll
  for (int r = 0; r < 8; ++r) {
    C[(size_t)(rb + r) * N + col0]      = acc0[r];
    C[(size_t)(rb + r) * N + col0 + 16] = acc1[r];
  }
}

// ---------- activation/bias + convert to bf16, zero-pad rows >= rows ----------
// mode: 0 = identity, 1 = ELU, 2 = ReLU
__global__ void k_prep_act(const float* __restrict__ src, const float* __restrict__ bias,
                           unsigned short* __restrict__ dst, int rows, int cols,
                           int mpad, int mode) {
  int i = blockIdx.x * blockDim.x + threadIdx.x;
  if (i >= mpad * cols) return;
  int r = i / cols, c = i - r * cols;
  float v = 0.f;
  if (r < rows) {
    v = src[(size_t)r * cols + c];
    if (bias) v += bias[c];
    if (mode == 1) v = (v > 0.f) ? v : (__expf(v) - 1.f);
    else if (mode == 2) v = fmaxf(v, 0.f);
  }
  dst[i] = f2bf(v);
}

__global__ void k_fill_u32(unsigned int* __restrict__ p, unsigned int v, int n) {
  int i = blockIdx.x * blockDim.x + threadIdx.x;
  if (i < n) p[i] = v;
}

// ---------- per-node attention coefficients ----------
__global__ void k_att(const float* __restrict__ H, const float* __restrict__ a_s,
                      const float* __restrict__ a_d, float* __restrict__ als,
                      float* __restrict__ ald, int nn, int heads, int ch) {
  int t = blockIdx.x * blockDim.x + threadIdx.x;
  if (t >= nn * heads) return;
  int n = t / heads, h = t - n * heads;
  const float* hp = H + (size_t)n * heads * ch + (size_t)h * ch;
  const float* sp = a_s + h * ch;
  const float* dp = a_d + h * ch;
  float s = 0.f, d = 0.f;
  for (int c = 0; c < ch; ++c) { float v = hp[c]; s += v * sp[c]; d += v * dp[c]; }
  als[t] = s; ald[t] = d;
}

// ---------- edge pass 1: leaky-relu logits + segment max (uint atomicMax) ----------
__global__ void k_edge_logits(const int* __restrict__ ei, int E, int nn,
                              const float* __restrict__ als, const float* __restrict__ ald,
                              float* __restrict__ ebuf, unsigned int* __restrict__ nmax,
                              int heads) {
  int t = blockIdx.x * blockDim.x + threadIdx.x;
  if (t >= (E + nn) * heads) return;
  int e = t / heads, h = t - e * heads;
  int s, d;
  if (e < E) { s = ei[e]; d = ei[E + e]; } else { s = e - E; d = s; }
  float v = als[s * heads + h] + ald[d * heads + h];
  v = (v > 0.f) ? v : 0.2f * v;
  ebuf[t] = v;
  atomicMax(&nmax[d * heads + h], fenc(v));
}

// ---------- edge pass 2: exp(e - max) + segment sum ----------
__global__ void k_edge_exp(const int* __restrict__ ei, int E, int nn,
                           float* __restrict__ ebuf, const unsigned int* __restrict__ nmax,
                           float* __restrict__ denom, int heads) {
  int t = blockIdx.x * blockDim.x + threadIdx.x;
  if (t >= (E + nn) * heads) return;
  int e = t / heads, h = t - e * heads;
  int d = (e < E) ? ei[E + e] : (e - E);
  float ee = __expf(ebuf[t] - fdec(nmax[d * heads + h]));
  ebuf[t] = ee;
  atomicAdd(&denom[d * heads + h], ee);
}

// ---------- edge pass 3: one wave per edge; gather h[src]*alpha, scatter-add ----------
__global__ __launch_bounds__(256) void k_agg(const int* __restrict__ ei, int E, int nn,
                      const float* __restrict__ H, const float* __restrict__ ebuf,
                      const float* __restrict__ denom, float* __restrict__ O,
                      int heads, int ch) {
  int gid  = blockIdx.x * blockDim.x + threadIdx.x;
  int e    = gid >> 5;
  int lane = gid & 31;
  if (e >= E + nn) return;
  int s, d;
  if (e < E) { s = ei[e]; d = ei[E + e]; } else { s = e - E; d = s; }
  const int cat = heads * ch;
  const float* hs = H + (size_t)s * cat;
  float* od       = O + (size_t)d * cat;
  __builtin_prefetch(hs + lane, 0, 3);   // global_prefetch_b8 on the gather row
  for (int c = lane; c < cat; c += 32) {
    int h   = c / ch;
    float a = ebuf[(size_t)e * heads + h] / (denom[d * heads + h] + 1e-16f);
    atomicAdd(&od[c], hs[c] * a);
  }
}

// ---------- final bias + store fp32 output ----------
__global__ void k_bias_out(const float* __restrict__ H, const float* __restrict__ b,
                           float* __restrict__ out, int rows, int cols) {
  int i = blockIdx.x * blockDim.x + threadIdx.x;
  if (i >= rows * cols) return;
  out[i] = H[i] + b[i % cols];
}

// =====================================================================
extern "C" void kernel_launch(void* const* d_in, const int* in_sizes, int n_in,
                              void* d_out, int out_size, void* d_ws, size_t ws_size,
                              hipStream_t stream) {
  const float* x     = (const float*)d_in[0];
  const int*   ei    = (const int*)d_in[1];
  const float* W1    = (const float*)d_in[2];
  const float* asrc1 = (const float*)d_in[3];
  const float* adst1 = (const float*)d_in[4];
  const float* b1    = (const float*)d_in[5];
  const float* W2    = (const float*)d_in[6];
  const float* asrc2 = (const float*)d_in[7];
  const float* adst2 = (const float*)d_in[8];
  const float* b2    = (const float*)d_in[9];
  const float* W3    = (const float*)d_in[10];
  const float* asrc3 = (const float*)d_in[11];
  const float* adst3 = (const float*)d_in[12];
  const float* b3    = (const float*)d_in[13];
  const float* M1    = (const float*)d_in[14];
  const float* mb1   = (const float*)d_in[15];
  const float* M2    = (const float*)d_in[16];
  const float* mb2   = (const float*)d_in[17];
  const float* M3    = (const float*)d_in[18];
  const float* mb3   = (const float*)d_in[19];

  const int NN   = in_sizes[0] / 256;                 // 50000 nodes
  const int E    = in_sizes[1] / 2;                   // 800000 edges
  const int Mpad = ((NN + BM - 1) / BM) * BM;         // 50048 (mult of 128)

  // ---- carve workspace (all re-initialized every call) ----
  char* p = (char*)d_ws;
  auto carve = [&](size_t bytes) -> char* {
    char* r = p; p += (bytes + 255) & ~(size_t)255; return r;
  };
  unsigned short* Abf  = (unsigned short*)carve((size_t)Mpad * 512 * 2);
  float*          Hf   = (float*)carve((size_t)Mpad * 512 * 4);
  float*          Of   = (float*)carve((size_t)Mpad * 512 * 4);
  float*          als  = (float*)carve((size_t)NN * 4 * 4);
  float*          ald  = (float*)carve((size_t)NN * 4 * 4);
  unsigned int*   nmax = (unsigned int*)carve((size_t)NN * 4 * 4);
  float*          denom= (float*)carve((size_t)NN * 4 * 4);
  float*          ebuf = (float*)carve((size_t)(E + NN) * 4 * 4);
  unsigned short* W1b  = (unsigned short*)carve((size_t)256 * 512 * 2);
  unsigned short* W2b  = (unsigned short*)carve((size_t)512 * 512 * 2);
  unsigned short* W3b  = (unsigned short*)carve((size_t)512 * 256 * 2);
  unsigned short* M1b  = (unsigned short*)carve((size_t)256 * 256 * 2);
  unsigned short* M2b  = (unsigned short*)carve((size_t)256 * 128 * 2);
  unsigned short* M3b  = (unsigned short*)carve((size_t)128 * 256 * 2);
  (void)ws_size; (void)n_in; (void)out_size;

  auto pack = [&](const float* w, unsigned short* o, int K, int Nc) {
    int nd = (K * Nc) >> 1;
    k_pack_w<<<(nd + 255) / 256, 256, 0, stream>>>(w, o, K, Nc);
  };
  pack(W1, W1b, 256, 512); pack(W2, W2b, 512, 512); pack(W3, W3b, 512, 256);
  pack(M1, M1b, 256, 256); pack(M2, M2b, 256, 128); pack(M3, M3b, 128, 256);

  auto gemm = [&](const unsigned short* Wb, int K, int Nc) {
    k_gemm_bf16<<<dim3(Mpad / BM, Nc / BN), 256, 0, stream>>>(Abf, Wb, Hf, K, Nc);
  };
  auto gat_layer = [&](const unsigned short* Wb, int K, int Nc,
                       const float* a_s, const float* a_d, int heads, int ch) {
    gemm(Wb, K, Nc);
    k_att<<<(NN * heads + 255) / 256, 256, 0, stream>>>(Hf, a_s, a_d, als, ald, NN, heads, ch);
    k_fill_u32<<<(NN * heads + 255) / 256, 256, 0, stream>>>(nmax, 0u, NN * heads);
    k_fill_u32<<<(NN * heads + 255) / 256, 256, 0, stream>>>((unsigned int*)denom, 0u, NN * heads);
    k_fill_u32<<<(Mpad * Nc + 255) / 256, 256, 0, stream>>>((unsigned int*)Of, 0u, Mpad * Nc);
    int totE = (E + NN) * heads;
    k_edge_logits<<<(totE + 255) / 256, 256, 0, stream>>>(ei, E, NN, als, ald, ebuf, nmax, heads);
    k_edge_exp<<<(totE + 255) / 256, 256, 0, stream>>>(ei, E, NN, ebuf, nmax, denom, heads);
    long aggT = (long)(E + NN) * 32;
    k_agg<<<(int)((aggT + 255) / 256), 256, 0, stream>>>(ei, E, NN, Hf, ebuf, denom, Of, heads, ch);
  };

  // ---- GAT layer 1: 256 -> 4x128 concat, ELU ----
  k_prep_act<<<(Mpad * 256 + 255) / 256, 256, 0, stream>>>(x, nullptr, Abf, NN, 256, Mpad, 0);
  gat_layer(W1b, 256, 512, asrc1, adst1, 4, 128);
  k_prep_act<<<(Mpad * 512 + 255) / 256, 256, 0, stream>>>(Of, b1, Abf, NN, 512, Mpad, 1);

  // ---- GAT layer 2: 512 -> 4x128 concat, ELU ----
  gat_layer(W2b, 512, 512, asrc2, adst2, 4, 128);
  k_prep_act<<<(Mpad * 512 + 255) / 256, 256, 0, stream>>>(Of, b2, Abf, NN, 512, Mpad, 1);

  // ---- GAT layer 3: 512 -> 256, 1 head (mean over 1 head == identity) ----
  gat_layer(W3b, 512, 256, asrc3, adst3, 1, 256);
  k_prep_act<<<(Mpad * 256 + 255) / 256, 256, 0, stream>>>(Of, b3, Abf, NN, 256, Mpad, 0);

  // ---- response MLP: 256 -> 256 -> 128 -> 256 ----
  gemm(M1b, 256, 256);
  k_prep_act<<<(Mpad * 256 + 255) / 256, 256, 0, stream>>>(Hf, mb1, Abf, NN, 256, Mpad, 2);
  gemm(M2b, 256, 128);
  k_prep_act<<<(Mpad * 128 + 255) / 256, 256, 0, stream>>>(Hf, mb2, Abf, NN, 128, Mpad, 2);
  gemm(M3b, 128, 256);
  k_bias_out<<<(NN * 256 + 255) / 256, 256, 0, stream>>>(Hf, mb3, (float*)d_out, NN, 256);
}
